// ResNeXtBlock_A_28698971472118
// MI455X (gfx1250) — matrix operand order repaired and checked by reading, not compiled, not run on hardware
//
#include <hip/hip_runtime.h>

// ---------------------------------------------------------------------------
// ResNeXt block A on MI455X (gfx1250): bf16 WMMA for both 1x1 GEMMs,
// packed-bf16 VALU (v_pk_fma_bf16) for BN1+ReLU activation generation,
// plain VALU for the tiny 3x3 grouped conv. BN/ReLU affines fully folded.
// HBM floor ~40MB @ 23.3TB/s ~= 1.7us; T1/T2 intermediates stay in 192MB L2.
// ---------------------------------------------------------------------------

#define G_   32
#define B_   16
#define H_   56
#define W_   56
#define CIN_ 128
#define BF_  4
#define COUT_ 256
#define NPIX1 (B_ * H_ * W_)      // 50176
#define NPIX2 (B_ * 28 * 28)      // 12544
#define EPS_ 1e-3f

typedef __attribute__((ext_vector_type(16))) __bf16 v16bf;
typedef __attribute__((ext_vector_type(2)))  __bf16 bf16x2;
typedef __attribute__((ext_vector_type(8)))  float  v8f;
typedef int vi4_ __attribute__((vector_size(16)));
typedef __attribute__((address_space(1))) vi4_* gas1_t;   // global v4i32*
typedef __attribute__((address_space(3))) vi4_* las3_t;   // LDS v4i32*

union V16BF { v16bf v; unsigned int u[8]; uint4 q[2]; };
union V8F   { v8f  v; float f[8]; };

__device__ __forceinline__ unsigned int bf16pair(float a, float b) {
    unsigned int ua = __float_as_uint(a), ub = __float_as_uint(b);
    ua = (ua + 0x7FFFu + ((ua >> 16) & 1u)) >> 16;   // RNE f32 -> bf16
    ub = (ub + 0x7FFFu + ((ub >> 16) & 1u)) >> 16;
    return (ua & 0xFFFFu) | (ub << 16);
}

// packed bf16 BN+ReLU: relu(x*s + t) on 2 channels -> v_pk_fma_bf16 + v_pk_max
__device__ __forceinline__ unsigned int act2(unsigned int xu, unsigned int su,
                                             unsigned int tu) {
    bf16x2 xv = __builtin_bit_cast(bf16x2, xu);
    bf16x2 sv = __builtin_bit_cast(bf16x2, su);
    bf16x2 tv = __builtin_bit_cast(bf16x2, tu);
    bf16x2 r = __builtin_elementwise_fma(xv, sv, tv);
    bf16x2 z = {(__bf16)0.0f, (__bf16)0.0f};
    r = __builtin_elementwise_max(r, z);
    return __builtin_bit_cast(unsigned int, r);
}

// workspace layout (32-bit units)
#define WS_S1BF  0            // 2048 u32: [g][c/2] bf16 scale pairs
#define WS_T1BF  2048         // 2048 u32: [g][c/2] bf16 shift pairs
#define WS_S2    4096         // 128 f32
#define WS_T2P   4224         // 128 f32
#define WS_S3    4352         // 128 f32
#define WS_T3P   4480         // 128 f32
#define WS_BSUM  4608         // 256 f32
#define WS_W1T   4864         // 8192 u32 : [g][f][c/2] bf16 pairs
#define WS_W2T   13056        // 16384 u32: [o][k/2]    bf16 pairs
#define WS_T1BUF 29696        // NPIX1*128 f32
#define WS_T2BF  (29696 + NPIX1 * 128)   // NPIX2*64 u32: [pix][ch/2] bf16 pairs

// ---------------------------------------------------------------------------
// k0: fold all BN affines (bf16-packed for BN1), sum final biases,
//     transpose+convert W1/W2 to bf16 pairs matching the WMMA B layout.
// ---------------------------------------------------------------------------
__global__ __launch_bounds__(256) void k0_setup(
    const float* __restrict__ g1, const float* __restrict__ be1,
    const float* __restrict__ m1, const float* __restrict__ v1,
    const float* __restrict__ W1, const float* __restrict__ b1,
    const float* __restrict__ g2, const float* __restrict__ be2,
    const float* __restrict__ m2, const float* __restrict__ v2,
    const float* __restrict__ b3,
    const float* __restrict__ g3, const float* __restrict__ be3,
    const float* __restrict__ m3, const float* __restrict__ v3,
    const float* __restrict__ W2, const float* __restrict__ b2,
    float* __restrict__ ws)
{
    const int t = threadIdx.x;
    unsigned int* wsu = (unsigned int*)ws;
    // BN1 scale/shift, packed as bf16 channel-pairs: [g][c/2]
    for (int i = t; i < G_ * CIN_ / 2; i += 256) {
        int c0 = i * 2, c1 = c0 + 1;
        float s0 = g1[c0] * __frsqrt_rn(v1[c0] + EPS_);
        float s1 = g1[c1] * __frsqrt_rn(v1[c1] + EPS_);
        wsu[WS_S1BF + i] = bf16pair(s0, s1);
        wsu[WS_T1BF + i] = bf16pair(be1[c0] - m1[c0] * s0, be1[c1] - m1[c1] * s1);
    }
    // BN2 folded with conv1 bias, BN3 folded with conv3 bias (ch = g*4+f)
    for (int i = t; i < G_ * BF_; i += 256) {
        float s2 = g2[i] * __frsqrt_rn(v2[i] + EPS_);
        ws[WS_S2 + i]  = s2;
        ws[WS_T2P + i] = s2 * b1[i] + (be2[i] - m2[i] * s2);
        float s3 = g3[i] * __frsqrt_rn(v3[i] + EPS_);
        ws[WS_S3 + i]  = s3;
        ws[WS_T3P + i] = s3 * b3[i] + (be3[i] - m3[i] * s3);
    }
    // bias sum over groups for the final 1x1
    for (int o = t; o < COUT_; o += 256) {
        float s = 0.f;
        for (int g = 0; g < G_; ++g) s += b2[g * COUT_ + o];
        ws[WS_BSUM + o] = s;
    }
    // W1 -> [g][f][c/2] bf16 pairs (K-pairs contiguous for B fragments)
    unsigned int* w1t = wsu + WS_W1T;
    for (int i = t; i < G_ * BF_ * (CIN_ / 2); i += 256) {
        int g = i >> 8, r = i & 255, f = r >> 6, c = (r & 63) * 2;
        w1t[i] = bf16pair(W1[g * (CIN_ * BF_) + c * BF_ + f],
                          W1[g * (CIN_ * BF_) + (c + 1) * BF_ + f]);
    }
    // W2 (flat [k=g*4+f][o]) -> [o][k/2] bf16 pairs
    unsigned int* w2t = wsu + WS_W2T;
    for (int i = t; i < COUT_ * (G_ * BF_ / 2); i += 256) {
        int o = i >> 6, k = (i & 63) * 2;
        w2t[i] = bf16pair(W2[k * COUT_ + o], W2[(k + 1) * COUT_ + o]);
    }
}

// ---------------------------------------------------------------------------
// k1: conv1 over all 32 groups as padded-N bf16 WMMA; fused BN1/ReLU (packed
//     bf16) in, fused BN2/ReLU out. 64 pixels/block, 8 waves (wave32).
// NOTE: B columns 4..15 carry garbage (duplicated group weights); since WMMA
// columns are independent and raise no exceptions, the garbage only lands in
// C columns the epilogue never stores -> no masking needed.
// ---------------------------------------------------------------------------
__global__ __launch_bounds__(256) void k1_conv1(
    const float* __restrict__ x, const float* __restrict__ ws,
    float* __restrict__ t1buf)
{
    __shared__ __align__(16) unsigned int xs2[64 * 64];   // 16 KB, swizzled bf16 pairs
    __shared__ __align__(16) unsigned int slds[2048];     // 8 KB BN1 scale pairs
    __shared__ __align__(16) unsigned int tlds[2048];     // 8 KB BN1 shift pairs
    const int t = threadIdx.x;
    const int pixBase = blockIdx.x * 64;
    const unsigned int* wsu = (const unsigned int*)ws;

    // stage x tile as bf16 pairs (coalesced b64 global reads, swizzled stores)
    #pragma unroll
    for (int i = 0; i < 16; ++i) {
        int lin = i * 256 + t;
        int p = lin >> 6, pc = lin & 63;
        const float* xp = x + (pixBase + p) * CIN_ + pc * 2;
        xs2[p * 64 + (pc ^ ((p & 15) << 2))] = bf16pair(xp[0], xp[1]);
    }
    // stage BN1 scale/shift tables (pure copy): async DMA to LDS if available
#if __has_builtin(__builtin_amdgcn_global_load_async_to_lds_b128)
    {
        #pragma unroll
        for (int i = 0; i < 2; ++i) {
            int off = (i * 256 + t) * 4;
            __builtin_amdgcn_global_load_async_to_lds_b128(
                (gas1_t)(void*)(wsu + WS_S1BF + off),
                (las3_t)(void*)(slds + off), 0, 0);
            __builtin_amdgcn_global_load_async_to_lds_b128(
                (gas1_t)(void*)(wsu + WS_T1BF + off),
                (las3_t)(void*)(tlds + off), 0, 0);
        }
#if __has_builtin(__builtin_amdgcn_s_wait_asynccnt)
        __builtin_amdgcn_s_wait_asynccnt(0);
#else
        asm volatile("s_wait_asynccnt 0" ::: "memory");
#endif
    }
#else
    #pragma unroll
    for (int i = 0; i < 8; ++i) {
        int idx = i * 256 + t;
        slds[idx] = wsu[WS_S1BF + idx];
        tlds[idx] = wsu[WS_T1BF + idx];
    }
#endif
    __syncthreads();

    const int lane = t & 31, wid = t >> 5;
    const int sub = wid & 3;             // 16-pixel subtile
    const int g0 = (wid >> 2) * 16;      // group half
    const int m = lane & 15, hi = lane >> 4;
    const int sw = m << 2;
    const int row = (sub * 16 + m) * 64;
    const unsigned int* w1t = wsu + WS_W1T;

    for (int g = g0; g < g0 + 16; ++g) {
        __builtin_prefetch(&w1t[(g + 1 < 32 ? g + 1 : g) * 256], 0, 0);
        V8F acc;
        #pragma unroll
        for (int r = 0; r < 8; ++r) acc.f[r] = 0.f;

        #pragma unroll
        for (int kc = 0; kc < 4; ++kc) {
            // ISA 16-bit A 16x32 layout: channel pairs pc0..pc0+3, pc0+8..+11
            const int pc0 = kc * 16 + hi * 4, pc1 = pc0 + 8;
            uint4 xa = *(const uint4*)&xs2[row + (pc0 ^ sw)];
            uint4 xb = *(const uint4*)&xs2[row + (pc1 ^ sw)];
            uint4 sa = *(const uint4*)&slds[g * 64 + pc0];
            uint4 sb = *(const uint4*)&slds[g * 64 + pc1];
            uint4 ta = *(const uint4*)&tlds[g * 64 + pc0];
            uint4 tb = *(const uint4*)&tlds[g * 64 + pc1];
            V16BF A;
            A.u[0] = act2(xa.x, sa.x, ta.x);
            A.u[1] = act2(xa.y, sa.y, ta.y);
            A.u[2] = act2(xa.z, sa.z, ta.z);
            A.u[3] = act2(xa.w, sa.w, ta.w);
            A.u[4] = act2(xb.x, sb.x, tb.x);
            A.u[5] = act2(xb.y, sb.y, tb.y);
            A.u[6] = act2(xb.z, sb.z, tb.z);
            A.u[7] = act2(xb.w, sb.w, tb.w);
            // ISA 16-bit B 32x16 layout; pairs boff..+3 and boff+8..+11 -> 2x b128
            const int boff = g * 256 + (m & 3) * 64 + kc * 16 + hi * 4;
            V16BF Bm;
            Bm.q[0] = *(const uint4*)&w1t[boff];
            Bm.q[1] = *(const uint4*)&w1t[boff + 8];
            acc.v = __builtin_amdgcn_wmma_f32_16x16x32_bf16(
                false, A.v, false, Bm.v, (short)0, acc.v, false, false);
        }
        // epilogue: cols 0..3 valid -> ch = g*4+f, fused BN2+ReLU
        if (m < 4) {
            int ch = g * 4 + m;
            float s2 = ws[WS_S2 + ch], t2 = ws[WS_T2P + ch];
            #pragma unroll
            for (int r = 0; r < 8; ++r) {
                int M = r + hi * 8;
                t1buf[(pixBase + sub * 16 + M) * 128 + ch] =
                    fmaxf(fmaf(acc.f[r], s2, t2), 0.f);
            }
        }
    }
}

// ---------------------------------------------------------------------------
// k2: 3x3 stride-2 grouped conv (4->4 per group) + fused BN3/ReLU (VALU);
//     emits T2 as packed bf16 channel-pairs (WMMA-ready for k3).
// ---------------------------------------------------------------------------
__global__ __launch_bounds__(256) void k2_conv3(
    const float* __restrict__ t1buf, const float* __restrict__ W3,
    const float* __restrict__ ws, unsigned int* __restrict__ t2bf)
{
    int idx = blockIdx.x * 256 + threadIdx.x;      // NPIX2 * G threads
    int g = idx & 31;
    int pix = idx >> 5;
    int b = pix / 784, rem = pix % 784;
    int oy = rem / 28, ox = rem % 28;

    float acc[4] = {0.f, 0.f, 0.f, 0.f};
    const float* w3g = W3 + g * 144;
    #pragma unroll
    for (int ky = 0; ky < 3; ++ky) {
        int iy = oy * 2 - 1 + ky;
        if (iy < 0 || iy >= H_) continue;
        #pragma unroll
        for (int kx = 0; kx < 3; ++kx) {
            int ix = ox * 2 - 1 + kx;
            if (ix < 0 || ix >= W_) continue;
            const float* tin = t1buf + ((b * H_ + iy) * W_ + ix) * 128 + g * 4;
            const float* wk = w3g + (ky * 3 + kx) * 16;
            #pragma unroll
            for (int fi = 0; fi < 4; ++fi) {
                float tv = tin[fi];
                #pragma unroll
                for (int fo = 0; fo < 4; ++fo)
                    acc[fo] = fmaf(tv, wk[fi * 4 + fo], acc[fo]);
            }
        }
    }
    float z[4];
    #pragma unroll
    for (int fo = 0; fo < 4; ++fo) {
        int ch = g * 4 + fo;
        z[fo] = fmaxf(fmaf(acc[fo], ws[WS_S3 + ch], ws[WS_T3P + ch]), 0.f);
    }
    uint2 pk;
    pk.x = bf16pair(z[0], z[1]);
    pk.y = bf16pair(z[2], z[3]);
    *(uint2*)&t2bf[pix * 64 + g * 2] = pk;       // one b64 store
}

// ---------------------------------------------------------------------------
// k3: [12544,128]x[128,256] bf16 WMMA + summed bias + strided-x residual.
//     A fragments load directly as packed bf16 (2x global b128, no cvt).
// ---------------------------------------------------------------------------
__global__ __launch_bounds__(256) void k3_conv2(
    const unsigned int* __restrict__ t2bf, const float* __restrict__ x,
    const float* __restrict__ ws, float* __restrict__ out)
{
    __shared__ float resbuf[32];
    const int t = threadIdx.x;
    const int pixBase = blockIdx.x * 32;
    if (t < 32) resbuf[t] = 0.f;
    __syncthreads();
    {   // residual: channel-sum of x[:, ::2, ::2, :], 8 threads per pixel
        int p = t >> 3, part = t & 7;
        int opix = pixBase + p;
        int b = opix / 784, rem = opix % 784;
        int oy = rem / 28, ox = rem % 28;
        const float* xp = x + ((b * H_ + oy * 2) * W_ + ox * 2) * CIN_ + part * 16;
        float s = 0.f;
        #pragma unroll
        for (int j = 0; j < 16; ++j) s += xp[j];
        atomicAdd(&resbuf[p], s);
    }
    __syncthreads();

    const int lane = t & 31, wid = t >> 5;
    const int msub = wid & 1;            // 16-pixel subtile
    const int quarter = wid >> 1;        // 64 output channels
    const int m = lane & 15, hi = lane >> 4;
    const unsigned int* w2t = (const unsigned int*)ws + WS_W2T;

    V8F C[4];
    #pragma unroll
    for (int nt = 0; nt < 4; ++nt)
        #pragma unroll
        for (int r = 0; r < 8; ++r) C[nt].f[r] = 0.f;

    const int pix = pixBase + msub * 16 + m;
    #pragma unroll
    for (int kc = 0; kc < 4; ++kc) {
        // same ISA A layout as k1: pairs pc0..pc0+3 and pc0+8..pc0+11
        const int pc0 = kc * 16 + hi * 4;
        V16BF A;
        A.q[0] = *(const uint4*)&t2bf[pix * 64 + pc0];
        A.q[1] = *(const uint4*)&t2bf[pix * 64 + pc0 + 8];
        #pragma unroll
        for (int nt = 0; nt < 4; ++nt) {
            int o = quarter * 64 + nt * 16 + m;
            const int boff = o * 64 + kc * 16 + hi * 4;   // u32 boff..+3 and boff+8..+11
            V16BF Bm;
            Bm.q[0] = *(const uint4*)&w2t[boff];
            Bm.q[1] = *(const uint4*)&w2t[boff + 8];
            C[nt].v = __builtin_amdgcn_wmma_f32_16x16x32_bf16(
                false, A.v, false, Bm.v, (short)0, C[nt].v, false, false);
        }
    }
    #pragma unroll
    for (int nt = 0; nt < 4; ++nt) {
        int o = quarter * 64 + nt * 16 + m;
        float bsum = ws[WS_BSUM + o];
        #pragma unroll
        for (int r = 0; r < 8; ++r) {
            int M = r + hi * 8;
            int spix = pixBase + msub * 16 + M;
            out[spix * COUT_ + o] = C[nt].f[r] + bsum + resbuf[msub * 16 + M];
        }
    }
}

// ---------------------------------------------------------------------------
extern "C" void kernel_launch(void* const* d_in, const int* in_sizes, int n_in,
                              void* d_out, int out_size, void* d_ws, size_t ws_size,
                              hipStream_t stream) {
    (void)in_sizes; (void)n_in; (void)out_size; (void)ws_size;
    const float* x   = (const float*)d_in[0];
    const float* g1  = (const float*)d_in[1];
    const float* be1 = (const float*)d_in[2];
    const float* m1  = (const float*)d_in[3];
    const float* v1  = (const float*)d_in[4];
    const float* W1  = (const float*)d_in[5];
    const float* b1  = (const float*)d_in[6];
    const float* g2  = (const float*)d_in[7];
    const float* be2 = (const float*)d_in[8];
    const float* m2  = (const float*)d_in[9];
    const float* v2  = (const float*)d_in[10];
    const float* W3  = (const float*)d_in[11];
    const float* b3  = (const float*)d_in[12];
    const float* g3  = (const float*)d_in[13];
    const float* be3 = (const float*)d_in[14];
    const float* m3  = (const float*)d_in[15];
    const float* v3  = (const float*)d_in[16];
    const float* W2  = (const float*)d_in[17];
    const float* b2  = (const float*)d_in[18];
    float* ws  = (float*)d_ws;
    float* out = (float*)d_out;

    k0_setup<<<1, 256, 0, stream>>>(g1, be1, m1, v1, W1, b1, g2, be2, m2, v2,
                                    b3, g3, be3, m3, v3, W2, b2, ws);
    k1_conv1<<<NPIX1 / 64, 256, 0, stream>>>(x, ws, ws + WS_T1BUF);
    k2_conv3<<<(NPIX2 * G_) / 256, 256, 0, stream>>>(
        ws + WS_T1BUF, W3, ws, (unsigned int*)ws + WS_T2BF);
    k3_conv2<<<NPIX2 / 32, 256, 0, stream>>>(
        (const unsigned int*)ws + WS_T2BF, x, ws, out);
}
